// KronRelationNets_64707977282199
// MI455X (gfx1250) — matrix operand order repaired and checked by reading, not compiled, not run on hardware
//
#include <hip/hip_runtime.h>
#include <hip/hip_bf16.h>
#include <math.h>

typedef _Float16 half8 __attribute__((ext_vector_type(8)));
typedef _Float16 v16h  __attribute__((ext_vector_type(16)));
typedef float    v8f   __attribute__((ext_vector_type(8)));

#define EPS_BN 1e-5f
#define EPS_L2 1e-12f

// ---------------- workspace (d_ws) layout, bytes ----------------
#define WS_W0H   0          // f16 [64][64]        layer0 weight, row=o, col=c
#define WS_W1H   8192       // f16 [9][64][128]    layer1 weight per tap; c: 0-63=xf, 64-127=yf
#define WS_W1K   155648     // f32 [64][9]         folded layer1 weight for kron map
#define WS_W1K2  157952     // f32 [64][9]         folded layer1 weight for kron_ map
#define WS_SU    160256     // f32 s0[64],u0[64],s1[64],u1[64],s2[64],u2[64]
#define WS_TOTAL 161792

// ---------------- LDS layout, bytes (regions reused across stages) -----
#define LDS_POOLX  0        // f32 [64][100]  pooled query     (stage 1-2)
#define LDS_POOLY  25600    // f32 [64][100]  pooled support   (stage 1-2)
#define LDS_S      0        // f32 [112][112] kron similarity  (stage 3-4)
#define LDS_ZF     0        // f16 [100][128] layer0 out [p][c] (stage 5-6)
#define LDS_OUT1   25600    // f32 [64][64]   layer1 out [o][p] (stage 6-7)
#define LDS_P1     41984    // f32 [64][16]   layer1 pooled [c][4x4]
#define LDS_TMP    46080    // f32 [256]
#define LDS_F      47104    // f32 [64]
#define LDS_H      47360    // f32 [8]
#define LDS_XH     51200    // f16 [112][64]  normalized x, [p][c], rows 100-111 zero
#define LDS_YH     65536    // f16 [112][64]  normalized y
#define LDS_W0     79872    // f16 [64][64]   staged layer0 weight
#define LDS_W1K    88064    // f32 [64][9]
#define LDS_W1K2   90368    // f32 [64][9]
#define LDS_KMAX   92672    // f32 [128]  kron  (rowmax of S)
#define LDS_K2MAX  93184    // f32 [128]  kron_ (colmax of S)
#define LDS_SU     93696    // f32 [384]  s0,u0,s1,u1,s2,u2
#define LDS_TOTAL  95232

#define WMMA_F16(a, b, c) \
  __builtin_amdgcn_wmma_f32_16x16x32_f16(false, (a), false, (b), (short)0, (c), false, false)

// A fragment (M16 x K32, f16): lane->row; halves 0-15 = k0+16*(lane>>4)+0..15.
// Source stored row-major [row][k], ldk halves per row.
__device__ __forceinline__ v16h load_a_frag(const _Float16* base, int ldk,
                                            int row0, int k0, int lane) {
  const _Float16* p = base + (row0 + (lane & 15)) * ldk + k0 + ((lane >> 4) << 4);
  half8 c0 = *(const half8*)p;
  half8 c1 = *(const half8*)(p + 8);
  return __builtin_shufflevector(c0, c1, 0, 1, 2, 3, 4, 5, 6, 7,
                                 8, 9, 10, 11, 12, 13, 14, 15);
}

// B fragment (K32 x N16, f16): lane->column n, 16 contiguous halves at
// k0 + 16*(lane>>4). Source stored row-major [n][k].
__device__ __forceinline__ v16h load_b_frag(const _Float16* base, int ldk,
                                            int n0, int k0, int lane) {
  return *(const v16h*)(base + (n0 + (lane & 15)) * ldk + k0 + ((lane >> 4) << 4));
}

// D/C layout: lane 0-15 -> N=lane, VGPR r -> M=r; lane 16-31 -> M=r+8.
__device__ __forceinline__ void layer1_epilogue(v8f acc, int mt, int lane, int p,
                                                const float* km, const float* k2,
                                                const float* W1K, const float* W1K2,
                                                const float* s1v, const float* u1v,
                                                float* Out1) {
  const int ob = mt * 16 + ((lane >> 4) << 3);
#pragma unroll
  for (int r = 0; r < 8; ++r) {
    const int o = ob + r;
    float v = acc[r];
#pragma unroll
    for (int tap = 0; tap < 9; ++tap)
      v += W1K[o * 9 + tap] * km[tap] + W1K2[o * 9 + tap] * k2[tap];
    Out1[o * 64 + p] = fmaxf(v * s1v[o] + u1v[o], 0.0f);
  }
}

// ---------------- one-time weight packing ----------------
__global__ __launch_bounds__(256) void kron_prep(
    const float* __restrict__ w0, const float* __restrict__ w1,
    const float* __restrict__ g0, const float* __restrict__ b0,
    const float* __restrict__ m0, const float* __restrict__ v0,
    const float* __restrict__ g1, const float* __restrict__ b1,
    const float* __restrict__ m1, const float* __restrict__ v1,
    const float* __restrict__ bc1,
    const float* __restrict__ g2, const float* __restrict__ b2,
    const float* __restrict__ m2, const float* __restrict__ v2,
    const float* __restrict__ bc2, char* __restrict__ ws) {
  _Float16* w0h = (_Float16*)(ws + WS_W0H);
  _Float16* w1h = (_Float16*)(ws + WS_W1H);
  float* w1k  = (float*)(ws + WS_W1K);
  float* w1k2 = (float*)(ws + WS_W1K2);
  float* su   = (float*)(ws + WS_SU);
  const int gtid = blockIdx.x * 256 + threadIdx.x;
  const int gsz  = gridDim.x * 256;

  for (int i = gtid; i < 64 * 64; i += gsz) w0h[i] = (_Float16)w0[i];

  for (int i = gtid; i < 9 * 64 * 128; i += gsz) {
    const int tap = i / (64 * 128);
    const int rem = i % (64 * 128);
    const int o = rem >> 7, c = rem & 127;
    const int ci = (c < 64) ? c : (c + 128);  // xf ch 0-63, yf ch 192-255
    w1h[i] = (_Float16)w1[(o * 256 + ci) * 9 + tap];
  }

  for (int i = gtid; i < 64 * 9; i += gsz) {
    const int o = i / 9, tap = i % 9;
    float a = 0.0f, b = 0.0f;
    for (int c = 0; c < 64; ++c) {
      a += w1[(o * 256 + 64 + c) * 9 + tap];    // kron block
      b += w1[(o * 256 + 128 + c) * 9 + tap];   // kron_ block
    }
    w1k[i] = a;
    w1k2[i] = b;
  }

  for (int i = gtid; i < 64; i += gsz) {
    const float s0 = g0[i] * rsqrtf(v0[i] + EPS_BN);
    su[i]       = s0;
    su[64 + i]  = b0[i] - m0[i] * s0;                 // layer0: no conv bias
    const float s1 = g1[i] * rsqrtf(v1[i] + EPS_BN);
    su[128 + i] = s1;
    su[192 + i] = (bc1[i] - m1[i]) * s1 + b1[i];
    const float s2 = g2[i] * rsqrtf(v2[i] + EPS_BN);
    su[256 + i] = s2;
    su[320 + i] = (bc2[i] - m2[i]) * s2 + b2[i];
  }
}

// ---------------- fully fused per-batch-element kernel ----------------
__global__ __launch_bounds__(256) void kron_relnet_fused(
    const float* __restrict__ support, const float* __restrict__ query,
    const _Float16* __restrict__ w0h_g, const _Float16* __restrict__ w1h_g,
    const float* __restrict__ w1k_g, const float* __restrict__ w1k2_g,
    const float* __restrict__ su_g, const float* __restrict__ w2,
    const float* __restrict__ wfc1, const float* __restrict__ bfc1,
    const float* __restrict__ wfc2, const float* __restrict__ bfc2,
    float* __restrict__ out) {
  extern __shared__ char smem[];
  const int tid  = threadIdx.x;
  const int lane = tid & 31;
  const int wave = tid >> 5;
  const int n    = blockIdx.x;

  float*    poolX = (float*)(smem + LDS_POOLX);
  float*    poolY = (float*)(smem + LDS_POOLY);
  _Float16* Xh    = (_Float16*)(smem + LDS_XH);
  _Float16* Yh    = (_Float16*)(smem + LDS_YH);
  _Float16* W0    = (_Float16*)(smem + LDS_W0);
  float*    W1K   = (float*)(smem + LDS_W1K);
  float*    W1K2  = (float*)(smem + LDS_W1K2);
  float*    KMAX  = (float*)(smem + LDS_KMAX);
  float*    K2MAX = (float*)(smem + LDS_K2MAX);
  float*    SU    = (float*)(smem + LDS_SU);

  const float* qbase = query   + (size_t)n * 64 * 400;
  const float* sbase = support + (size_t)n * 64 * 400;
  __builtin_prefetch(qbase + tid * 64, 0, 1);
  __builtin_prefetch(sbase + tid * 64, 0, 1);

  // ---- stage constants + 2x2 mean pool (HBM -> LDS) ----
  for (int i = tid; i < 64 * 64; i += 256) W0[i] = w0h_g[i];
  for (int i = tid; i < 576; i += 256) { W1K[i] = w1k_g[i]; W1K2[i] = w1k2_g[i]; }
  for (int i = tid; i < 384; i += 256) SU[i] = su_g[i];

  for (int i = tid; i < 6400; i += 256) {
    const int c = i / 100, p = i % 100, pi = p / 10, pj = p % 10;
    const float* q0 = qbase + (c * 20 + 2 * pi) * 20 + 2 * pj;
    const float* s0 = sbase + (c * 20 + 2 * pi) * 20 + 2 * pj;
    poolX[c * 100 + p] = 0.25f * (q0[0] + q0[1] + q0[20] + q0[21]);
    poolY[c * 100 + p] = 0.25f * (s0[0] + s0[1] + s0[20] + s0[21]);
  }
  __syncthreads();

  // ---- L2 normalize over channels; transpose to [pos][chan] f16 ----
  if (tid < 200) {
    const int p = (tid < 100) ? tid : (tid - 100);
    const float* src = (tid < 100) ? poolX : poolY;
    _Float16*   dst = (tid < 100) ? Xh : Yh;
    float ss = 0.0f;
    for (int c = 0; c < 64; ++c) { const float v = src[c * 100 + p]; ss += v * v; }
    const float inv = 1.0f / fmaxf(sqrtf(ss), EPS_L2);
    for (int c = 0; c < 64; ++c)
      dst[p * 64 + c] = (_Float16)(src[c * 100 + p] * inv);
  }
  for (int i = tid; i < 12 * 64; i += 256) {   // zero pad rows 100..111
    Xh[100 * 64 + i] = (_Float16)0.0f;
    Yh[100 * 64 + i] = (_Float16)0.0f;
  }
  __syncthreads();

  // ---- kron GEMM: S[p][q] = sum_c x[p][c] * y[q][c]  (112x112x64, WMMA) ----
  float* S = (float*)(smem + LDS_S);
  for (int t = wave; t < 49; t += 8) {
    const int pt = t / 7, qt = t % 7;
    v8f acc = {};
#pragma unroll
    for (int ks = 0; ks < 2; ++ks) {
      const v16h a = load_a_frag(Xh, 64, pt * 16, ks * 32, lane);
      const v16h b = load_b_frag(Yh, 64, qt * 16, ks * 32, lane);
      acc = WMMA_F16(a, b, acc);
    }
    const int qq = qt * 16 + (lane & 15);
    const int mb = pt * 16 + ((lane >> 4) << 3);
#pragma unroll
    for (int r = 0; r < 8; ++r) S[(mb + r) * 112 + qq] = acc[r];
  }
  __syncthreads();

  // ---- row / column maxes -> kron, kron_ ----
  if (tid < 100) {
    float m = -3.4e38f;
    for (int q = 0; q < 100; ++q) m = fmaxf(m, S[tid * 112 + q]);
    KMAX[tid] = m;
  } else if (tid < 200) {
    const int q = tid - 100;
    float m = -3.4e38f;
    for (int p = 0; p < 100; ++p) m = fmaxf(m, S[p * 112 + q]);
    K2MAX[q] = m;
  }
  __syncthreads();

  // ---- layer0: 1x1 conv (64x112x64 GEMM) + BN + ReLU, both streams ----
  _Float16* ZF = (_Float16*)(smem + LDS_ZF);   // [p][128]: 0-63 xf, 64-127 yf
  for (int job = wave; job < 56; job += 8) {
    const int st = job / 28, rem = job % 28, ot = rem / 7, pt = rem % 7;
    const _Float16* src = st ? Yh : Xh;
    v8f acc = {};
#pragma unroll
    for (int ks = 0; ks < 2; ++ks) {
      const v16h a = load_a_frag(W0, 64, ot * 16, ks * 32, lane);
      const v16h b = load_b_frag(src, 64, pt * 16, ks * 32, lane);
      acc = WMMA_F16(a, b, acc);
    }
    const int p = pt * 16 + (lane & 15);
    if (p < 100) {
      const int ob = ot * 16 + ((lane >> 4) << 3);
#pragma unroll
      for (int r = 0; r < 8; ++r) {
        const int o = ob + r;
        const float z = fmaxf(acc[r] * SU[o] + SU[64 + o], 0.0f);
        ZF[p * 128 + st * 64 + o] = (_Float16)z;
      }
    }
  }
  __syncthreads();

  // ---- layer1: 3x3 conv as 9-tap GEMM (M=64,N=64,K=9*128) + kron fold ----
  float* Out1 = (float*)(smem + LDS_OUT1);
  {
    const int nt = wave & 3;
    const int mp = (wave >> 2) << 1;            // this wave: m-tiles {mp, mp+1}
    const int p  = nt * 16 + (lane & 15);       // output position 0..63 (8x8)
    const int ii = p >> 3, jj = p & 7;
    v8f acc0 = {}, acc1 = {};
    for (int tap = 0; tap < 9; ++tap) {
      const int kh = tap / 3, kw = tap % 3;
      const int srow = (ii + kh) * 10 + (jj + kw);
      const _Float16* Bp = ZF + srow * 128 + ((lane >> 4) << 4);
      const _Float16* Ab = w1h_g + tap * 64 * 128;
#pragma unroll
      for (int ks = 0; ks < 4; ++ks) {
        const v16h b  = *(const v16h*)(Bp + ks * 32);
        const v16h a0 = load_a_frag(Ab, 128, mp * 16, ks * 32, lane);
        const v16h a1 = load_a_frag(Ab, 128, mp * 16 + 16, ks * 32, lane);
        acc0 = WMMA_F16(a0, b, acc0);
        acc1 = WMMA_F16(a1, b, acc1);
      }
    }
    float km[9], k2[9];
#pragma unroll
    for (int tap = 0; tap < 9; ++tap) {
      const int srow = (ii + tap / 3) * 10 + (jj + tap % 3);
      km[tap] = KMAX[srow];
      k2[tap] = K2MAX[srow];
    }
    layer1_epilogue(acc0, mp,     lane, p, km, k2, W1K, W1K2, SU + 128, SU + 192, Out1);
    layer1_epilogue(acc1, mp + 1, lane, p, km, k2, W1K, W1K2, SU + 128, SU + 192, Out1);
  }
  __syncthreads();

  // ---- maxpool 8x8 -> 4x4 ----
  float* P1 = (float*)(smem + LDS_P1);
  for (int i = tid; i < 64 * 16; i += 256) {
    const int o = i >> 4, cell = i & 15, pi2 = cell >> 2, pj2 = cell & 3;
    const float* row = Out1 + o * 64;
    const int b0 = (2 * pi2) * 8 + 2 * pj2;
    P1[i] = fmaxf(fmaxf(row[b0], row[b0 + 1]), fmaxf(row[b0 + 8], row[b0 + 9]));
  }
  __syncthreads();

  // ---- layer2: 3x3 conv 64->64 on 4x4 -> 2x2, VALU dot products ----
  float* tmp = (float*)(smem + LDS_TMP);
  {
    const int o = tid >> 2, q = tid & 3, qi = q >> 1, qj = q & 1;
    const float* wrow = w2 + o * 576;
    float acc = 0.0f;
    for (int c = 0; c < 64; ++c) {
      const float* pc = P1 + c * 16 + qi * 4 + qj;
      const float* wc = wrow + c * 9;
#pragma unroll
      for (int kh = 0; kh < 3; ++kh)
#pragma unroll
        for (int kw = 0; kw < 3; ++kw)
          acc += wc[kh * 3 + kw] * pc[kh * 4 + kw];
    }
    tmp[tid] = fmaxf(acc * SU[256 + o] + SU[320 + o], 0.0f);
  }
  __syncthreads();

  // ---- maxpool 2x2 -> 1, FC 64->8->1 ----
  float* F = (float*)(smem + LDS_F);
  float* H = (float*)(smem + LDS_H);
  if (tid < 64)
    F[tid] = fmaxf(fmaxf(tmp[4 * tid], tmp[4 * tid + 1]),
                   fmaxf(tmp[4 * tid + 2], tmp[4 * tid + 3]));
  __syncthreads();
  if (tid < 8) {
    float a = bfc1[tid];
    for (int o = 0; o < 64; ++o) a += F[o] * wfc1[tid * 64 + o];
    H[tid] = fmaxf(a, 0.0f);
  }
  __syncthreads();
  if (tid == 0) {
    float a = bfc2[0];
#pragma unroll
    for (int j = 0; j < 8; ++j) a += H[j] * wfc2[j];
    out[n] = a;
  }
}

extern "C" void kernel_launch(void* const* d_in, const int* in_sizes, int n_in,
                              void* d_out, int out_size, void* d_ws, size_t ws_size,
                              hipStream_t stream) {
  (void)n_in; (void)out_size; (void)ws_size;
  const float* support = (const float*)d_in[0];
  const float* query   = (const float*)d_in[1];
  const float* w0   = (const float*)d_in[2];
  const float* g0   = (const float*)d_in[3];
  const float* b0   = (const float*)d_in[4];
  const float* m0   = (const float*)d_in[5];
  const float* v0   = (const float*)d_in[6];
  const float* w1   = (const float*)d_in[7];
  const float* bc1  = (const float*)d_in[8];
  const float* g1   = (const float*)d_in[9];
  const float* b1   = (const float*)d_in[10];
  const float* m1   = (const float*)d_in[11];
  const float* v1   = (const float*)d_in[12];
  const float* w2   = (const float*)d_in[13];
  const float* bc2  = (const float*)d_in[14];
  const float* g2   = (const float*)d_in[15];
  const float* b2   = (const float*)d_in[16];
  const float* m2   = (const float*)d_in[17];
  const float* v2   = (const float*)d_in[18];
  const float* wfc1 = (const float*)d_in[19];
  const float* bfc1 = (const float*)d_in[20];
  const float* wfc2 = (const float*)d_in[21];
  const float* bfc2 = (const float*)d_in[22];

  const int B = in_sizes[0] / (64 * 20 * 20);
  char* ws = (char*)d_ws;

  kron_prep<<<64, 256, 0, stream>>>(w0, w1, g0, b0, m0, v0, g1, b1, m1, v1, bc1,
                                    g2, b2, m2, v2, bc2, ws);

  kron_relnet_fused<<<B, 256, LDS_TOTAL, stream>>>(
      support, query,
      (const _Float16*)(ws + WS_W0H), (const _Float16*)(ws + WS_W1H),
      (const float*)(ws + WS_W1K), (const float*)(ws + WS_W1K2),
      (const float*)(ws + WS_SU), w2, wfc1, bfc1, wfc2, bfc2, (float*)d_out);
}
